// Attention_57037165691576
// MI455X (gfx1250) — compile-verified, matrix operand order checked by reading, not run on hardware
//
#include <hip/hip_runtime.h>
#include <math.h>

// Problem sizes (fixed by the reference)
#define D_MODEL 1024
#define T_SEQ   2048
#define BATCH   2
#define NHEAD   16
#define DKEY    64
#define MROWS   (BATCH * T_SEQ)   // 4096
#define N_QKV   (3 * D_MODEL)     // 3072

typedef __attribute__((ext_vector_type(8)))  float  v8f;
typedef __attribute__((ext_vector_type(4)))  float  v4f;
typedef __attribute__((ext_vector_type(16))) __bf16 v16bf;
typedef __attribute__((ext_vector_type(8)))  __bf16 v8bf;

// Async global -> LDS copy, 16B per lane (ASYNCcnt-tracked).
#define ASYNC_B128(ldsaddr, gaddr)                                          \
  asm volatile("global_load_async_to_lds_b128 %0, %1, off"                  \
               :: "v"(ldsaddr), "v"(gaddr) : "memory")

#define WAIT_ASYNC(n) asm volatile("s_wait_asynccnt " #n ::: "memory")
#define WAIT_DS0()    asm volatile("s_wait_dscnt 0" ::: "memory")

static __device__ __forceinline__ uint32_t lds_addr(const void* p) {
  return (uint32_t)(uintptr_t)p;   // low 32 bits of the LDS aperture address
}
static __device__ __forceinline__ uint64_t g_addr(const void* p) {
  return (uint64_t)(uintptr_t)p;
}

// 16-element bf16 operand for 16x16x32 WMMA:
// chunk0 = K [g*8 .. g*8+7], chunk1 = K [16+g*8 .. 16+g*8+7]  (g = lane/16).
static __device__ __forceinline__ v16bf load_ab16(const __bf16* p0, const __bf16* p1) {
  v8bf lo = *(const v8bf*)p0;
  v8bf hi = *(const v8bf*)p1;
  v16bf r;
#pragma unroll
  for (int k = 0; k < 8; ++k) { r[k] = lo[k]; r[k + 8] = hi[k]; }
  return r;
}

static __device__ __forceinline__ v8f wmma_bf16(v16bf a, v16bf b, v8f c) {
  return __builtin_amdgcn_wmma_f32_16x16x32_bf16(false, a, false, b, (short)0, c,
                                                 false, false);
}

// ---------------------------------------------------------------------------
// Kernel 1: fp32 -> bf16, 8 elements/thread, b128 in / b128 out
// ---------------------------------------------------------------------------
__global__ void k_convert_bf16(const float* __restrict__ in, __bf16* __restrict__ out,
                               int n8) {
  int i = blockIdx.x * blockDim.x + threadIdx.x;
  int stride = gridDim.x * blockDim.x;
  for (; i < n8; i += stride) {
    v4f a = *(const v4f*)(in + (size_t)i * 8);
    v4f b = *(const v4f*)(in + (size_t)i * 8 + 4);
    v8bf o;
#pragma unroll
    for (int k = 0; k < 4; ++k) { o[k] = (__bf16)a[k]; o[4 + k] = (__bf16)b[k]; }
    *(v8bf*)(out + (size_t)i * 8) = o;
  }
}

// ---------------------------------------------------------------------------
// Projections: block = 256 threads (8 waves), 128M x 64N block tile,
// K-step 64, async double-buffered LDS staging, manual ping-pong so the
// accumulators stay pinned and each phase uses constant LDS offsets.
// ---------------------------------------------------------------------------
#define KSTEP 64

template <bool IS_QKV>
static __device__ __forceinline__ void gemm_body(
    const __bf16* __restrict__ Xsrc, const __bf16* __restrict__ Wsrc,
    const float* __restrict__ bias, int nTilesN,
    __bf16* __restrict__ Qb, __bf16* __restrict__ Kb, __bf16* __restrict__ Vt,
    float* __restrict__ Out) {
  __shared__ __align__(16) __bf16 Al[2][128 * KSTEP];  // 2 x 16 KiB
  __shared__ __align__(16) __bf16 Bl[2][64 * KSTEP];   // 2 x  8 KiB

  const int mBase = (int)(blockIdx.x / nTilesN) * 128;
  const int nBase = (int)(blockIdx.x % nTilesN) * 64;
  const int tid   = threadIdx.x;
  const int wv    = tid >> 5;
  const int lane  = tid & 31;
  const int lr = lane & 15, lh = lane >> 4;

  // ---- async stage of one K-step (6 x b128 per thread) ----
  auto issue = [&](int s) {
    const int buf = s & 1;
    const int kk  = s * KSTEP;
#pragma unroll
    for (int r = 0; r < 4; ++r) {            // A: 128x64 -> 1024 16B chunks
      const int c   = tid + 256 * r;
      const int row = c >> 3, c16 = c & 7;
      ASYNC_B128(lds_addr(&Al[buf][row * KSTEP + c16 * 8]),
                 g_addr(Xsrc + (size_t)(mBase + row) * D_MODEL + kk + c16 * 8));
    }
#pragma unroll
    for (int r = 0; r < 2; ++r) {            // B: 64x64 -> 512 16B chunks
      const int c   = tid + 256 * r;
      const int row = c >> 3, c16 = c & 7;
      ASYNC_B128(lds_addr(&Bl[buf][row * KSTEP + c16 * 8]),
                 g_addr(Wsrc + (size_t)(nBase + row) * D_MODEL + kk + c16 * 8));
    }
  };

  v8f acc0 = {}, acc1 = {}, acc2 = {}, acc3 = {};

  // Batch operand loads first (one ds clause + one wait), then 4 WMMAs.
  auto compute = [&](const __bf16* Abuf, const __bf16* Bbuf) {
#pragma unroll
    for (int half = 0; half < 2; ++half) {
      const __bf16* apl = Abuf + (wv * 16 + lr) * KSTEP + half * 32 + lh * 8;
      const __bf16* b0p = Bbuf + (0 * 16 + lr) * KSTEP + half * 32 + lh * 8;
      const __bf16* b1p = Bbuf + (1 * 16 + lr) * KSTEP + half * 32 + lh * 8;
      const __bf16* b2p = Bbuf + (2 * 16 + lr) * KSTEP + half * 32 + lh * 8;
      const __bf16* b3p = Bbuf + (3 * 16 + lr) * KSTEP + half * 32 + lh * 8;
      v16bf a  = load_ab16(apl, apl + 16);
      v16bf b0 = load_ab16(b0p, b0p + 16);
      v16bf b1 = load_ab16(b1p, b1p + 16);
      v16bf b2 = load_ab16(b2p, b2p + 16);
      v16bf b3 = load_ab16(b3p, b3p + 16);
      acc0 = wmma_bf16(a, b0, acc0);
      acc1 = wmma_bf16(a, b1, acc1);
      acc2 = wmma_bf16(a, b2, acc2);
      acc3 = wmma_bf16(a, b3, acc3);
    }
  };

  const int NSTAGE = D_MODEL / KSTEP;        // 16 (even)
  issue(0);
  for (int s = 0; s < NSTAGE; s += 2) {
    if (s + 1 < NSTAGE) { issue(s + 1); WAIT_ASYNC(6); } else { WAIT_ASYNC(0); }
    __syncthreads();
    compute(&Al[0][0], &Bl[0][0]);
    __syncthreads();
    if (s + 2 < NSTAGE) { issue(s + 2); WAIT_ASYNC(6); } else { WAIT_ASYNC(0); }
    __syncthreads();
    compute(&Al[1][0], &Bl[1][0]);
    __syncthreads();
  }

  // ---- epilogue ----
  v8f accs[4] = {acc0, acc1, acc2, acc3};
#pragma unroll
  for (int t = 0; t < 4; ++t) {
#pragma unroll
    for (int i = 0; i < 8; ++i) {
      const int row = mBase + wv * 16 + i + 8 * lh;
      const int col = nBase + t * 16 + lr;
      const float val = accs[t][i] + bias[col];
      if (IS_QKV) {
        const __bf16 bv = (__bf16)val;
        const int region = col >> 10;              // 0=Q 1=K 2=V
        const int dfull  = col & (D_MODEL - 1);
        const int hh = dfull >> 6;
        const int dd = dfull & (DKEY - 1);
        const int bb = row >> 11;
        const int tt = row & (T_SEQ - 1);
        const size_t bh = (size_t)bb * NHEAD + hh;
        if (region == 0)      Qb[(bh * T_SEQ + tt) * DKEY + dd] = bv;
        else if (region == 1) Kb[(bh * T_SEQ + tt) * DKEY + dd] = bv;
        else                  Vt[(bh * DKEY + dd) * T_SEQ + tt] = bv;
      } else {
        Out[(size_t)row * D_MODEL + col] = val;
      }
    }
  }
}

__global__ __launch_bounds__(256) void k_gemm_qkv(
    const __bf16* __restrict__ Xb, const __bf16* __restrict__ Wb,
    const float* __restrict__ bias,
    __bf16* __restrict__ Qb, __bf16* __restrict__ Kb, __bf16* __restrict__ Vt) {
  gemm_body<true>(Xb, Wb, bias, N_QKV / 64, Qb, Kb, Vt, nullptr);
}

__global__ __launch_bounds__(256) void k_gemm_out(
    const __bf16* __restrict__ Cb, const __bf16* __restrict__ Wb,
    const float* __restrict__ bias, float* __restrict__ Out) {
  gemm_body<false>(Cb, Wb, bias, D_MODEL / 64, nullptr, nullptr, nullptr, Out);
}

// ---------------------------------------------------------------------------
// Flash attention: 4 waves share one (b,h); 16-query tile per wave; keys in
// chunks of 32 staged once per block via async copies (double-buffered,
// ping-pong).  Online softmax in base-2; P staged through per-wave LDS to
// turn the WMMA C-layout into an A-layout.
// ---------------------------------------------------------------------------
#define ATT_WAVES 4
__global__ __launch_bounds__(32 * ATT_WAVES) void k_flash_attn(
    const __bf16* __restrict__ Qb, const __bf16* __restrict__ Kb,
    const __bf16* __restrict__ Vt, __bf16* __restrict__ Ctx) {
  __shared__ __align__(16) __bf16 Kl[2][32 * DKEY];       // 2 x 4 KiB
  __shared__ __align__(16) __bf16 Vl[2][DKEY * 32];       // 2 x 4 KiB
  __shared__ __align__(16) __bf16 Pl[ATT_WAVES][16 * 32]; // 4 x 1 KiB

  const int tid  = threadIdx.x;
  const int wloc = tid >> 5;
  const int lane = tid & 31;
  const int wid  = blockIdx.x * ATT_WAVES + wloc;
  const int bh   = wid >> 7;            // block-uniform
  const int q0   = (wid & 127) * 16;
  const int lr = lane & 15, lh = lane >> 4;

  const float C1 = 0.125f * 1.4426950408889634f;   // log2(e)/sqrt(dk)

  // ---- async stage of one 32-key chunk (4 x b128 per thread) ----
  auto issue = [&](int s) {
    const int buf = s & 1;
    const int j0  = s * 32;
#pragma unroll
    for (int r = 0; r < 2; ++r) {        // K chunk: 32x64 -> 256 16B chunks
      const int c   = tid + 128 * r;
      const int row = c >> 3, c16 = c & 7;
      ASYNC_B128(lds_addr(&Kl[buf][row * DKEY + c16 * 8]),
                 g_addr(Kb + ((size_t)bh * T_SEQ + j0 + row) * DKEY + c16 * 8));
    }
#pragma unroll
    for (int r = 0; r < 2; ++r) {        // V^T chunk: 64x32 -> 256 16B chunks
      const int c   = tid + 128 * r;
      const int row = c >> 2, c16 = c & 3;
      ASYNC_B128(lds_addr(&Vl[buf][row * 32 + c16 * 8]),
                 g_addr(Vt + ((size_t)bh * DKEY + row) * T_SEQ + j0 + c16 * 8));
    }
  };

  // Q tile resident for the whole pass.
  const __bf16* qp = Qb + ((size_t)bh * T_SEQ + q0 + lr) * DKEY + lh * 8;
  const v16bf a0 = load_ab16(qp, qp + 16);        // d =  0..31
  const v16bf a1 = load_ab16(qp + 32, qp + 48);   // d = 32..63

  v8f accO0 = {}, accO1 = {}, accO2 = {}, accO3 = {};
  float mrow[8], lrow[8];
#pragma unroll
  for (int i = 0; i < 8; ++i) { mrow[i] = -__builtin_inff(); lrow[i] = 0.0f; }

  __bf16* const pl = &Pl[wloc][0];

  auto attn_step = [&](const __bf16* Kbuf, const __bf16* Vbuf) {
    // ---- S = Q . K^T : batch the 4 B operands, then 4 WMMAs ----
    const __bf16* k0 = Kbuf + lr * DKEY + lh * 8;
    const __bf16* k1 = Kbuf + (16 + lr) * DKEY + lh * 8;
    v16bf bk00 = load_ab16(k0, k0 + 16);
    v16bf bk01 = load_ab16(k0 + 32, k0 + 48);
    v16bf bk10 = load_ab16(k1, k1 + 16);
    v16bf bk11 = load_ab16(k1 + 32, k1 + 48);
    v8f s0 = {}, s1 = {};
    s0 = wmma_bf16(a0, bk00, s0);
    s0 = wmma_bf16(a1, bk01, s0);
    s1 = wmma_bf16(a0, bk10, s1);
    s1 = wmma_bf16(a1, bk11, s1);

    // ---- online softmax over the 32 new logits per row ----
    float alpha[8];
#pragma unroll
    for (int i = 0; i < 8; ++i) {
      float t0 = s0[i] * C1;
      float t1 = s1[i] * C1;
      float mx = fmaxf(t0, t1);
      mx = fmaxf(mx, __shfl_xor(mx, 1, 32));
      mx = fmaxf(mx, __shfl_xor(mx, 2, 32));
      mx = fmaxf(mx, __shfl_xor(mx, 4, 32));
      mx = fmaxf(mx, __shfl_xor(mx, 8, 32));
      const float mnew = fmaxf(mrow[i], mx);
      const float p0 = exp2f(t0 - mnew);
      const float p1 = exp2f(t1 - mnew);
      float rs = p0 + p1;
      rs += __shfl_xor(rs, 1, 32);
      rs += __shfl_xor(rs, 2, 32);
      rs += __shfl_xor(rs, 4, 32);
      rs += __shfl_xor(rs, 8, 32);
      alpha[i] = exp2f(mrow[i] - mnew);
      lrow[i]  = lrow[i] * alpha[i] + rs;
      mrow[i]  = mnew;
      const int prow = i + 8 * lh;
      pl[prow * 32 + lr]      = (__bf16)p0;
      pl[prow * 32 + 16 + lr] = (__bf16)p1;
    }

#pragma unroll
    for (int i = 0; i < 8; ++i) {
      accO0[i] *= alpha[i];
      accO1[i] *= alpha[i];
      accO2[i] *= alpha[i];
      accO3[i] *= alpha[i];
    }

    WAIT_DS0();   // same-wave LDS RAW on the P tile

    // ---- O += P . V : batch operand loads, then 4 WMMAs ----
    const __bf16* pp  = pl + lr * 32 + lh * 8;
    const __bf16* v0p = Vbuf + (0 * 16 + lr) * 32 + lh * 8;
    const __bf16* v1p = Vbuf + (1 * 16 + lr) * 32 + lh * 8;
    const __bf16* v2p = Vbuf + (2 * 16 + lr) * 32 + lh * 8;
    const __bf16* v3p = Vbuf + (3 * 16 + lr) * 32 + lh * 8;
    v16bf pa = load_ab16(pp, pp + 16);
    v16bf bv0 = load_ab16(v0p, v0p + 16);
    v16bf bv1 = load_ab16(v1p, v1p + 16);
    v16bf bv2 = load_ab16(v2p, v2p + 16);
    v16bf bv3 = load_ab16(v3p, v3p + 16);
    accO0 = wmma_bf16(pa, bv0, accO0);
    accO1 = wmma_bf16(pa, bv1, accO1);
    accO2 = wmma_bf16(pa, bv2, accO2);
    accO3 = wmma_bf16(pa, bv3, accO3);
  };

  const int NSTAGE = T_SEQ / 32;                  // 64 (even)
  issue(0);
  for (int s = 0; s < NSTAGE; s += 2) {
    if (s + 1 < NSTAGE) { issue(s + 1); WAIT_ASYNC(4); } else { WAIT_ASYNC(0); }
    __syncthreads();
    attn_step(&Kl[0][0], &Vl[0][0]);
    __syncthreads();
    if (s + 2 < NSTAGE) { issue(s + 2); WAIT_ASYNC(4); } else { WAIT_ASYNC(0); }
    __syncthreads();
    attn_step(&Kl[1][0], &Vl[1][0]);
    __syncthreads();
  }

  // ---- normalize and write context [B, T, H*dk] bf16 ----
  const int bb = bh >> 4;
  const int hh = bh & (NHEAD - 1);
  v8f accs[4] = {accO0, accO1, accO2, accO3};
#pragma unroll
  for (int i = 0; i < 8; ++i) {
    const float inv = 1.0f / lrow[i];
    const int q = q0 + i + 8 * lh;
#pragma unroll
    for (int t = 0; t < 4; ++t) {
      const int d = hh * DKEY + t * 16 + lr;
      Ctx[((size_t)bb * T_SEQ + q) * D_MODEL + d] = (__bf16)(accs[t][i] * inv);
    }
  }
}

// ---------------------------------------------------------------------------
// Host-side launcher
// ---------------------------------------------------------------------------
extern "C" void kernel_launch(void* const* d_in, const int* in_sizes, int n_in,
                              void* d_out, int out_size, void* d_ws, size_t ws_size,
                              hipStream_t stream) {
  (void)in_sizes; (void)n_in; (void)out_size; (void)ws_size;
  const float* x    = (const float*)d_in[0];
  const float* Wqkv = (const float*)d_in[1];
  const float* bqkv = (const float*)d_in[2];
  const float* Wo   = (const float*)d_in[3];
  const float* bo   = (const float*)d_in[4];
  float* out = (float*)d_out;

  char* ws = (char*)d_ws;
  __bf16* Xb  = (__bf16*)ws;  ws += (size_t)MROWS * D_MODEL * 2;
  __bf16* Wqb = (__bf16*)ws;  ws += (size_t)N_QKV * D_MODEL * 2;
  __bf16* Wob = (__bf16*)ws;  ws += (size_t)D_MODEL * D_MODEL * 2;
  __bf16* Qb  = (__bf16*)ws;  ws += (size_t)BATCH * NHEAD * T_SEQ * DKEY * 2;
  __bf16* Kb  = (__bf16*)ws;  ws += (size_t)BATCH * NHEAD * T_SEQ * DKEY * 2;
  __bf16* Vt  = (__bf16*)ws;  ws += (size_t)BATCH * NHEAD * DKEY * T_SEQ * 2;
  __bf16* Ctx = (__bf16*)ws;  ws += (size_t)MROWS * D_MODEL * 2;

  int n8 = MROWS * D_MODEL / 8;
  k_convert_bf16<<<(n8 + 255) / 256, 256, 0, stream>>>(x, Xb, n8);
  n8 = N_QKV * D_MODEL / 8;
  k_convert_bf16<<<(n8 + 255) / 256, 256, 0, stream>>>(Wqkv, Wqb, n8);
  n8 = D_MODEL * D_MODEL / 8;
  k_convert_bf16<<<(n8 + 255) / 256, 256, 0, stream>>>(Wo, Wob, n8);

  // QKV projection: (4096/128) * (3072/64) = 1536 blocks
  k_gemm_qkv<<<(MROWS / 128) * (N_QKV / 64), 256, 0, stream>>>(Xb, Wqb, bqkv,
                                                               Qb, Kb, Vt);

  // Flash attention: 32 (b,h) * 128 q-tiles / 4 waves = 1024 blocks
  k_flash_attn<<<BATCH * NHEAD * (T_SEQ / 16) / ATT_WAVES, 32 * ATT_WAVES, 0,
                 stream>>>(Qb, Kb, Vt, Ctx);

  // Output projection: 32 * 16 = 512 blocks
  k_gemm_out<<<(MROWS / 128) * (D_MODEL / 64), 256, 0, stream>>>(Ctx, Wob, bo, out);
}
